// MultiHeadAttentionRoPE_51196010168601
// MI455X (gfx1250) — matrix operand order repaired and checked by reading, not compiled, not run on hardware
//
#include <hip/hip_runtime.h>
#include <hip/hip_bf16.h>
#include <math.h>

// ---------------------------------------------------------------------------
// MultiHeadAttention + RoPE for MI455X (gfx1250, wave32, WMMA).
// All matmuls use v_wmma_f32_16x16x32_bf16 (bf16 operands, f32 accum).
// Problem constants from the reference: B=2, S=2048, D=1024, H=16, hd=64.
// Workspace requirement: ~97 MB (see layout in kernel_launch).
//
// Perf model (MI455X): dominated by the mandatory 536 MB f32 qk output write
// plus bounded qk re-reads (stats + softmax*V) ~= 2.1 GB @ 23.3 TB/s.
// 70 GFLOP of bf16 WMMA is negligible against the matrix-core ceiling, so
// tiling maximizes stream efficiency: 32x64 tiles per wave (8 independent
// WMMA chains -> loads overlap WMMA), global_prefetch for streamed operands.
// ---------------------------------------------------------------------------

typedef __bf16 bf16_t;
typedef __bf16 bf16x16 __attribute__((ext_vector_type(16)));
typedef float  f32x8   __attribute__((ext_vector_type(8)));

// ---- fragment helpers (CDNA5 16-bit WMMA VGPR layouts, ISA 7.12.2) --------

// A-matrix 16x32: lane (row=lane&15, half=lane>>4) holds
// K = {half*8 .. half*8+7} (vec elems 0..7) and {16+half*8 .. +7} (elems 8..15).
// `base` must point at &A[row*lda + half*8 + k0]; both chunks 16B-aligned.
static __device__ inline bf16x16 load_frag_a(const bf16_t* base) {
  union { bf16x16 v; uint4 u[2]; } f;
  f.u[0] = *reinterpret_cast<const uint4*>(base);
  f.u[1] = *reinterpret_cast<const uint4*>(base + 16);
  return f.v;
}

// B-matrix 32x16 (B = W^T): lane n (=lane&15) holds column n of B over
// K = {half*8 .. half*8+7} (elems 0..7) and {16+half*8 .. +7} (elems 8..15).
static __device__ inline bf16x16 load_frag_b(const bf16_t* base) {
  union { bf16x16 v; uint4 u[2]; } f;
  f.u[0] = *reinterpret_cast<const uint4*>(base);
  f.u[1] = *reinterpret_cast<const uint4*>(base + 16);
  return f.v;
}

static __device__ inline f32x8 wmma_bf16(bf16x16 a, bf16x16 b, f32x8 c) {
  // (neg_a, A, neg_b, B, c_mod, C, reuse_a, reuse_b)
  return __builtin_amdgcn_wmma_f32_16x16x32_bf16(false, a, false, b,
                                                 (short)0, c, false, false);
}

// ---- elementwise converts --------------------------------------------------

__global__ void k_f32_to_bf16(const float* __restrict__ in,
                              bf16_t* __restrict__ out, size_t n) {
  size_t i = (size_t)blockIdx.x * blockDim.x + threadIdx.x;
  if (i < n) out[i] = (bf16_t)in[i];
}

// ---- generic GEMM: C[M,N] = A[M,K] @ W[N,K]^T + bias, f32 out -------------
// block = 128 threads (4 waves); each wave -> one 32x64 output tile
// (2 A row-tiles x 4 B col-tiles = 8 independent WMMA chains per K-step).
__global__ void k_gemm_xwt(const bf16_t* __restrict__ A,
                           const bf16_t* __restrict__ W,
                           const float* __restrict__ bias,
                           float* __restrict__ C,
                           int M, int N, int K) {
  const int lane  = threadIdx.x & 31;
  const int wave  = threadIdx.x >> 5;
  const int n_base = (blockIdx.x * (blockDim.x >> 5) + wave) * 64;
  const int m0     = blockIdx.y * 32;
  if (n_base >= N || m0 >= M) return;   // wave-uniform: EXEC stays all-1s
  const int row  = lane & 15;
  const int half = lane >> 4;

  const bf16_t* ap0 = A + (size_t)(m0 + row) * K + half * 8;
  const bf16_t* ap1 = A + (size_t)(m0 + 16 + row) * K + half * 8;
  const bf16_t* wp[4];
#pragma unroll
  for (int t = 0; t < 4; ++t)
    wp[t] = W + (size_t)(n_base + t * 16 + row) * K + half * 8;

  f32x8 acc0[4] = {};
  f32x8 acc1[4] = {};
  // K loop in 128-element chunks: prefetch 2 chunks ahead, 4 WMMA-steps/chunk.
  for (int kc = 0; kc < K; kc += 128) {
    if (kc + 256 < K) {
      __builtin_prefetch(ap0 + kc + 256, 0, 3);
      __builtin_prefetch(ap1 + kc + 256, 0, 3);
#pragma unroll
      for (int t = 0; t < 4; ++t) __builtin_prefetch(wp[t] + kc + 256, 0, 3);
    }
#pragma unroll
    for (int ks = 0; ks < 128; ks += 32) {
      const int k0 = kc + ks;
      bf16x16 a0 = load_frag_a(ap0 + k0);
      bf16x16 a1 = load_frag_a(ap1 + k0);
#pragma unroll
      for (int t = 0; t < 4; ++t) {
        bf16x16 b = load_frag_b(wp[t] + k0);
        acc0[t] = wmma_bf16(a0, b, acc0[t]);
        acc1[t] = wmma_bf16(a1, b, acc1[t]);
      }
    }
  }
#pragma unroll
  for (int t = 0; t < 4; ++t) {
    const int n  = n_base + t * 16 + row;
    const float bb = bias ? bias[n] : 0.0f;
#pragma unroll
    for (int v = 0; v < 8; ++v) {
      C[(size_t)(m0 + v + 8 * half) * N + n]      = acc0[t][v] + bb;
      C[(size_t)(m0 + 16 + v + 8 * half) * N + n] = acc1[t][v] + bb;
    }
  }
}

// ---- RoPE + scale: f32 [B,S,D] -> bf16 [B,H,S,hd] --------------------------
__global__ void k_rope(const float* __restrict__ q_pre,
                       const float* __restrict__ k_pre,
                       const int* __restrict__ pos_ids,
                       bf16_t* __restrict__ q_r, bf16_t* __restrict__ k_r,
                       int B, int S, int H, int hd, float scale) {
  int tid = blockIdx.x * blockDim.x + threadIdx.x;
  int total = B * H * S * hd;
  if (tid >= total) return;
  int d = tid % hd;
  int s = (tid / hd) % S;
  int h = (tid / (hd * S)) % H;
  int b = tid / (hd * S * H);
  int half_hd = hd >> 1;
  int i = d & (half_hd - 1);
  float pos = (float)pos_ids[b * S + s];
  float inv = powf(10000.0f, -(float)(2 * i) / (float)hd);
  float ang = pos * inv;
  float c = cosf(ang), sn = sinf(ang);
  size_t src = ((size_t)b * S + s) * (size_t)(H * hd) + (size_t)h * hd;
  float qv = q_pre[src + d] * scale;
  float kv = k_pre[src + d] * scale;
  float qo = (d < half_hd) ? -q_pre[src + d + half_hd] * scale
                           :  q_pre[src + d - half_hd] * scale;
  float ko = (d < half_hd) ? -k_pre[src + d + half_hd] * scale
                           :  k_pre[src + d - half_hd] * scale;
  size_t dst = (((size_t)b * H + h) * S + s) * hd + d;
  q_r[dst] = (bf16_t)(qv * c + qo * sn);
  k_r[dst] = (bf16_t)(kv * c + ko * sn);
}

// ---- V transpose: f32 [B,S,D] -> bf16 vT [B,H,hd,S] ------------------------
__global__ void k_transpose_v(const float* __restrict__ v_pre,
                              bf16_t* __restrict__ vT,
                              int B, int S, int H, int hd) {
  int tid = blockIdx.x * blockDim.x + threadIdx.x;
  int total = B * H * hd * S;
  if (tid >= total) return;
  int s = tid % S;
  int d = (tid / S) % hd;
  int h = (tid / (S * hd)) % H;
  int b = tid / (S * hd * H);
  vT[tid] = (bf16_t)v_pre[((size_t)b * S + s) * (size_t)(H * hd) +
                          (size_t)h * hd + d];
}

// ---- qk = q_r @ k_r^T per (b,h), masked -inf, f32 straight into d_out ------
// Bound by the f32 tile stores; K=64 -> fully unrolled 2 K-steps.
__global__ void k_gemm_qk(const bf16_t* __restrict__ q_r,
                          const bf16_t* __restrict__ k_r,
                          const int* __restrict__ mask,
                          float* __restrict__ qk,
                          int S, int hd, int H) {
  const int z = blockIdx.z;           // b*H + h
  const int b = z / H;
  const int lane  = threadIdx.x & 31;
  const int wave  = threadIdx.x >> 5;
  const int n_base = (blockIdx.x * (blockDim.x >> 5) + wave) * 64;
  const int m0     = blockIdx.y * 16;
  const int row  = lane & 15;
  const int half = lane >> 4;

  const bf16_t* A = q_r + (size_t)z * S * hd;
  const bf16_t* W = k_r + (size_t)z * S * hd;
  const bf16_t* ap = A + (size_t)(m0 + row) * hd + half * 8;
  const bf16_t* wp[4];
#pragma unroll
  for (int t = 0; t < 4; ++t)
    wp[t] = W + (size_t)(n_base + t * 16 + row) * hd + half * 8;

  f32x8 acc[4] = {};
#pragma unroll
  for (int k0 = 0; k0 < 64; k0 += 32) {      // hd=64
    bf16x16 a = load_frag_a(ap + k0);
#pragma unroll
    for (int t = 0; t < 4; ++t)
      acc[t] = wmma_bf16(a, load_frag_b(wp[t] + k0), acc[t]);
  }
  float* C = qk + (size_t)z * S * S;
  const int* mrow = mask + b * S;
#pragma unroll
  for (int t = 0; t < 4; ++t) {
    const int n = n_base + t * 16 + row;
    const bool dead = (mrow[n] == 0);
#pragma unroll
    for (int v = 0; v < 8; ++v)
      C[(size_t)(m0 + v + 8 * half) * S + n] = dead ? -INFINITY : acc[t][v];
  }
}

// ---- per-row softmax stats (max, sum) over qk rows, one wave32 per row -----
__global__ void k_row_stats(const float* __restrict__ qk,
                            float2* __restrict__ stats, int S, int nrows) {
  const int r = blockIdx.x * (blockDim.x >> 5) + (threadIdx.x >> 5);
  if (r >= nrows) return;
  const int lane = threadIdx.x & 31;
  const float* p = qk + (size_t)r * S;
  float m = -INFINITY;
  for (int i = lane; i < S; i += 32) {
    __builtin_prefetch(p + i + 512, 0, 0);
    m = fmaxf(m, p[i]);
  }
#pragma unroll
  for (int o = 16; o >= 1; o >>= 1) m = fmaxf(m, __shfl_xor(m, o, 32));
  float sum = 0.0f;
  for (int i = lane; i < S; i += 32) sum += __expf(p[i] - m);
#pragma unroll
  for (int o = 16; o >= 1; o >>= 1) sum += __shfl_xor(sum, o, 32);
  if (lane == 0) stats[r] = make_float2(m, sum);
}

// ---- fused softmax + (w @ v): A-frag = bf16(exp(s-m)/l) built on the fly ---
// wave computes full 16 x hd(=64) output tile for one row-block of one (b,h).
__global__ void k_attn_wv(const float* __restrict__ qk,
                          const float2* __restrict__ stats,
                          const bf16_t* __restrict__ vT,   // [Z, hd, S]
                          bf16_t* __restrict__ wv,          // [B, S, H*hd]
                          int S, int H, int hd) {
  const int z = blockIdx.z;
  const int b = z / H, h = z % H;
  const int lane  = threadIdx.x & 31;
  const int wave  = threadIdx.x >> 5;
  const int m0    = blockIdx.y * ((blockDim.x >> 5) * 16) + wave * 16;
  const int row   = lane & 15;
  const int half  = lane >> 4;

  const float* qrow = qk + ((size_t)z * S + (m0 + row)) * S;
  const float2 st   = stats[(size_t)z * S + m0 + row];
  const float mrow  = st.x;
  const float rinv  = 1.0f / st.y;

  const bf16_t* vp[4];
#pragma unroll
  for (int t = 0; t < 4; ++t)
    vp[t] = vT + ((size_t)z * hd + t * 16 + row) * S + half * 8;

  f32x8 acc[4] = {};
  for (int k0 = 0; k0 < S; k0 += 32) {
    const float* p = qrow + k0 + half * 8;
    if (k0 + 128 < S) {
      __builtin_prefetch(p + 128, 0, 3);
#pragma unroll
      for (int t = 0; t < 4; ++t) __builtin_prefetch(vp[t] + k0 + 128, 0, 3);
    }
    bf16x16 a;
#pragma unroll
    for (int j = 0; j < 8; ++j) {
      a[j]     = (bf16_t)(__expf(p[j]      - mrow) * rinv);
      a[8 + j] = (bf16_t)(__expf(p[16 + j] - mrow) * rinv);
    }
#pragma unroll
    for (int t = 0; t < 4; ++t)
      acc[t] = wmma_bf16(a, load_frag_b(vp[t] + k0), acc[t]);
  }
  const int Dm = H * hd;
#pragma unroll
  for (int t = 0; t < 4; ++t) {
    const int d = t * 16 + row;
#pragma unroll
    for (int v = 0; v < 8; ++v)
      wv[((size_t)b * S + m0 + v + 8 * half) * Dm + (size_t)h * hd + d] =
          (bf16_t)acc[t][v];
  }
}

// ---------------------------------------------------------------------------

extern "C" void kernel_launch(void* const* d_in, const int* in_sizes, int n_in,
                              void* d_out, int out_size, void* d_ws,
                              size_t ws_size, hipStream_t stream) {
  const float* x    = (const float*)d_in[0];
  const int*   mask = (const int*)d_in[1];
  const int*   pos  = (const int*)d_in[2];
  // d_in[3] = n_head (device scalar; value fixed by reference = 16)
  const float* Wq = (const float*)d_in[4];
  const float* bq = (const float*)d_in[5];
  const float* Wk = (const float*)d_in[6];
  const float* Wv = (const float*)d_in[7];
  const float* bv = (const float*)d_in[8];
  const float* Wo = (const float*)d_in[9];
  const float* bo = (const float*)d_in[10];

  constexpr int B = 2, S = 2048, D = 1024, H = 16, HD = 64;
  constexpr int M = B * S;                 // 4096
  constexpr int Z = B * H;                 // 32
  const float scale = 0.35355339059327379f; // 64^-0.25 = 2^-1.5

  float* out = (float*)d_out;                       // [B,S,D] f32
  float* qk  = out + (size_t)B * S * D;             // [B,H,S,S] f32

  // ---- workspace layout (~97 MB) ----
  char* w = (char*)d_ws;
  auto alloc = [&](size_t bytes) {
    char* p = w;
    w += (bytes + 255) & ~(size_t)255;
    return p;
  };
  bf16_t* x_bf  = (bf16_t*)alloc((size_t)M * D * 2);
  bf16_t* Wq_bf = (bf16_t*)alloc((size_t)D * D * 2);
  bf16_t* Wk_bf = (bf16_t*)alloc((size_t)D * D * 2);
  bf16_t* Wv_bf = (bf16_t*)alloc((size_t)D * D * 2);
  bf16_t* Wo_bf = (bf16_t*)alloc((size_t)D * D * 2);
  float*  q_pre = (float*)alloc((size_t)M * D * 4);
  float*  k_pre = (float*)alloc((size_t)M * D * 4);
  float*  v_pre = (float*)alloc((size_t)M * D * 4);
  bf16_t* q_r   = (bf16_t*)alloc((size_t)Z * S * HD * 2);
  bf16_t* k_r   = (bf16_t*)alloc((size_t)Z * S * HD * 2);
  bf16_t* vT    = (bf16_t*)alloc((size_t)Z * HD * S * 2);
  bf16_t* wv_bf = (bf16_t*)alloc((size_t)M * D * 2);
  float2* stats = (float2*)alloc((size_t)Z * S * 8);
  (void)ws_size; (void)in_sizes; (void)n_in; (void)out_size;

  // 1) converts to bf16
  {
    size_t n = (size_t)M * D;
    k_f32_to_bf16<<<dim3((unsigned)((n + 255) / 256)), dim3(256), 0, stream>>>(x, x_bf, n);
    size_t nw = (size_t)D * D;
    unsigned gw = (unsigned)((nw + 255) / 256);
    k_f32_to_bf16<<<dim3(gw), dim3(256), 0, stream>>>(Wq, Wq_bf, nw);
    k_f32_to_bf16<<<dim3(gw), dim3(256), 0, stream>>>(Wk, Wk_bf, nw);
    k_f32_to_bf16<<<dim3(gw), dim3(256), 0, stream>>>(Wv, Wv_bf, nw);
    k_f32_to_bf16<<<dim3(gw), dim3(256), 0, stream>>>(Wo, Wo_bf, nw);
  }

  // 2) projections: q_pre = x@Wq^T + bq ; k_pre = x@Wk^T ; v_pre = x@Wv^T + bv
  {
    dim3 grid(D / 256, M / 32);   // 4 waves x 64 cols, 32 rows per block
    dim3 blk(128);
    k_gemm_xwt<<<grid, blk, 0, stream>>>(x_bf, Wq_bf, bq, q_pre, M, D, D);
    k_gemm_xwt<<<grid, blk, 0, stream>>>(x_bf, Wk_bf, nullptr, k_pre, M, D, D);
    k_gemm_xwt<<<grid, blk, 0, stream>>>(x_bf, Wv_bf, bv, v_pre, M, D, D);
  }

  // 3) RoPE + scale -> q_r/k_r [B,H,S,hd] bf16 ; V transpose -> vT [B,H,hd,S]
  {
    int total = B * H * S * HD;                 // 4.19M
    unsigned g = (unsigned)((total + 255) / 256);
    k_rope<<<dim3(g), dim3(256), 0, stream>>>(q_pre, k_pre, pos, q_r, k_r,
                                              B, S, H, HD, scale);
    k_transpose_v<<<dim3(g), dim3(256), 0, stream>>>(v_pre, vT, B, S, H, HD);
  }

  // 4) qk (masked) -> d_out qk slice
  {
    dim3 grid(S / 256, S / 16, Z);
    dim3 blk(128);
    k_gemm_qk<<<grid, blk, 0, stream>>>(q_r, k_r, mask, qk, S, HD, H);
  }

  // 5) per-row softmax stats
  {
    int nrows = Z * S;                          // 65536
    dim3 grid((unsigned)(nrows / 4));           // 4 waves (rows) per block
    dim3 blk(128);
    k_row_stats<<<grid, blk, 0, stream>>>(qk, stats, S, nrows);
  }

  // 6) fused softmax + w@v -> wv bf16 [B,S,D]
  {
    dim3 grid(1, S / 64, Z);                    // 4 waves x 16 rows per block
    dim3 blk(128);
    k_attn_wv<<<grid, blk, 0, stream>>>(qk, stats, vT, wv_bf, S, H, HD);
  }

  // 7) out = wv @ Wo^T + bo -> d_out out slice
  {
    dim3 grid(D / 256, M / 32);
    dim3 blk(128);
    k_gemm_xwt<<<grid, blk, 0, stream>>>(wv_bf, Wo_bf, bo, out, M, D, D);
  }
}
